// ScaledDotProductAttentionModule_1795296330325
// MI455X (gfx1250) — matrix operand order repaired and checked by reading, not compiled, not run on hardware
//
#include <hip/hip_runtime.h>
#include <hip/hip_bf16.h>
#include <math.h>
#include <type_traits>

// Problem constants from the reference.
#define B_ 2
#define H_ 16
#define S_ 2048
#define D_ 64
#define NHEADS (B_ * H_)

typedef __attribute__((ext_vector_type(16))) _Float16 v16h;
typedef __attribute__((ext_vector_type(8)))  _Float16 v8h;
typedef __attribute__((ext_vector_type(8)))  float    v8f;

#define WAVES_PER_BLOCK 8
#define QROWS 32                  // query rows per wave (two 16-row M tiles)

// v_permlane16_b32: same 16-lane permutation applied to both halves of the
// wave -- exactly an xor-shuffle that never crosses a 16-lane row.
__device__ __forceinline__ float permlane16_f(float x, unsigned lo, unsigned hi) {
    return __uint_as_float(__builtin_amdgcn_permlane16(
        __float_as_uint(x), __float_as_uint(x), lo, hi, false, false));
}
// Nibble tables for lane^1, lane^2, lane^4, lane^8 within a 16-lane row.
#define XOR1_LO 0x67452301u
#define XOR1_HI 0xEFCDAB89u
#define XOR2_LO 0x54761032u
#define XOR2_HI 0xDCFE98BAu
#define XOR4_LO 0x32107654u
#define XOR4_HI 0xBA98FEDCu
#define XOR8_LO 0xFEDCBA98u
#define XOR8_HI 0x76543210u

__device__ __forceinline__ float rowmax16(float x) {
    x = fmaxf(x, permlane16_f(x, XOR1_LO, XOR1_HI));
    x = fmaxf(x, permlane16_f(x, XOR2_LO, XOR2_HI));
    x = fmaxf(x, permlane16_f(x, XOR4_LO, XOR4_HI));
    x = fmaxf(x, permlane16_f(x, XOR8_LO, XOR8_HI));
    return x;
}

// ---------------------------------------------------------------------------
// Prep kernel: K -> f16 row-major cache, V -> f16 transposed cache [h][d][s].
// ---------------------------------------------------------------------------
__global__ __launch_bounds__(256)
void convert_kv_kernel(const float* __restrict__ K, const float* __restrict__ V,
                       _Float16* __restrict__ kh, _Float16* __restrict__ vt)
{
    const size_t i = (size_t)blockIdx.x * 256 + threadIdx.x;   // 0 .. NHEADS*S*D-1
    kh[i] = (_Float16)K[i];
    const size_t head = i / (D_ * S_);
    const size_t rem  = i % (D_ * S_);
    const size_t d    = rem / S_;
    const size_t s    = rem % S_;
    vt[i] = (_Float16)V[head * S_ * D_ + s * D_ + d];
}

// ---------------------------------------------------------------------------
// Flash attention, base-2 logit domain. One wave = 32 query rows; 18 WMMAs per
// 32-key block (8 QK^T + 8 PV + 2 row-sum). Causal mask only on the diagonal
// block. Row-max via v_permlane16 (VALU), row-sum via a ones-matrix WMMA.
// ---------------------------------------------------------------------------
template <bool F16CACHE>
__global__ __launch_bounds__(256)
void ScaledDotProductAttentionModule_1795296330325_kernel(
    const float* __restrict__ Q, const float* __restrict__ K,
    const float* __restrict__ V, const _Float16* __restrict__ Kh,
    const _Float16* __restrict__ Vt, float* __restrict__ Out)
{
    // Per-wave LDS scratch: P tile 32x32 f16 = 2 KB per wave (16 KB / block).
    __shared__ __align__(16) _Float16 lds_p[WAVES_PER_BLOCK][QROWS * 32];

    const int lane = threadIdx.x & 31;
    const int wave = threadIdx.x >> 5;
    const int half = lane >> 4;       // which 16-lane half
    const int ln   = lane & 15;       // lane within half

    const int task   = blockIdx.x * WAVES_PER_BLOCK + wave;   // 0..2047
    const int q_tile = task & (S_ / QROWS - 1);               // 0..63
    const int head   = task >> 6;                             // 0..31
    const int q0     = q_tile * QROWS;
    const size_t base = (size_t)head * S_ * D_;

    // 1/sqrt(64) * log2(e): base-2 logit domain -> raw v_exp_f32, no pre-mul.
    const float scale = 0.125f * 1.44269504088896340736f;

    // ---- Q as f16 A-frags: [Mtile u][d-chunk c], 16x32 each ----
    v16h qa[2][2];
#pragma unroll
    for (int u = 0; u < 2; ++u) {
        const float* qrow = Q + base + (size_t)(q0 + u * 16 + ln) * D_;
#pragma unroll
        for (int c = 0; c < 2; ++c) {
            const int d0 = c * 32 + half * 16;
#pragma unroll
            for (int i = 0; i < 8; ++i) {
                qa[u][c][i]     = (_Float16)(qrow[d0 + i]      * scale);
                qa[u][c][i + 8] = (_Float16)(qrow[d0 + 8 + i] * scale);
            }
        }
    }

    // Ones B-frag: row-sum of P as a WMMA (sum lands in C layout = l layout).
    v16h ones;
#pragma unroll
    for (int i = 0; i < 16; ++i) ones[i] = (_Float16)1.0f;

    // ---- Flash-attention state ----
    v8f o[2][4];
#pragma unroll
    for (int u = 0; u < 2; ++u)
#pragma unroll
        for (int f = 0; f < 4; ++f) o[u][f] = (v8f)0.0f;
    float mrow[2][8], lrow[2][8];
#pragma unroll
    for (int u = 0; u < 2; ++u)
#pragma unroll
        for (int r = 0; r < 8; ++r) { mrow[u][r] = -1e30f; lrow[u][r] = 0.0f; }

    _Float16* pw = &lds_p[wave][0];

    // ---------------- 32-key block body (mask only on diagonal) -------------
    auto block_body = [&](int j0, auto MASKED_) {
        constexpr bool MASKED = decltype(MASKED_)::value;

        if (F16CACHE && !MASKED) {
            // gfx1250 global_prefetch_b8 for the next key block
            __builtin_prefetch(Kh + base + (size_t)(j0 + 32 + ln) * D_, 0, 1);
            __builtin_prefetch(Vt + base + (size_t)(16 * half + ln) * S_ + j0 + 32, 0, 1);
        }

        // ---- S = Q * K^T : 4 score tiles s[u][t] ----
        v8f s[2][2];
#pragma unroll
        for (int t = 0; t < 2; ++t) {
            const int jt = j0 + 16 * t;
            v16h kb[2];
#pragma unroll
            for (int c = 0; c < 2; ++c) {
                if (F16CACHE) {
                    kb[c] = *(const v16h*)(Kh + base + (size_t)(jt + ln) * D_
                                           + c * 32 + half * 16);
                } else {
                    const float* kp = K + base + (size_t)(jt + ln) * D_
                                      + c * 32 + half * 16;
#pragma unroll
                    for (int i = 0; i < 16; ++i) kb[c][i] = (_Float16)kp[i];
                }
            }
#pragma unroll
            for (int u = 0; u < 2; ++u) {
                v8f acc = (v8f)0.0f;
                acc = __builtin_amdgcn_wmma_f32_16x16x32_f16(
                    false, qa[u][0], false, kb[0], (short)0, acc, false, false);
                acc = __builtin_amdgcn_wmma_f32_16x16x32_f16(
                    false, qa[u][1], false, kb[1], (short)0, acc, false, false);
                s[u][t] = acc;
            }
        }

        // ---- (mask +) online softmax per M tile, base-2 domain ----
        float alpha2[2][8];
#pragma unroll
        for (int u = 0; u < 2; ++u) {
            if (MASKED) {
                const int qrow_base = q0 + u * 16 + 8 * half;
#pragma unroll
                for (int t = 0; t < 2; ++t) {
                    const int j = j0 + 16 * t + ln;
#pragma unroll
                    for (int r = 0; r < 8; ++r)
                        s[u][t][r] = (j > qrow_base + r) ? -1e30f : s[u][t][r];
                }
            }
#pragma unroll
            for (int r = 0; r < 8; ++r) {
                // row max over 32 keys: VALU-only permlane16 butterfly
                const float mx = rowmax16(fmaxf(s[u][0][r], s[u][1][r]));
                const float mnew = fmaxf(mrow[u][r], mx);
                alpha2[u][r] = __builtin_amdgcn_exp2f(mrow[u][r] - mnew);
                mrow[u][r] = mnew;
            }
#pragma unroll
            for (int t = 0; t < 2; ++t)
#pragma unroll
                for (int r = 0; r < 8; ++r)
                    s[u][t][r] = __builtin_amdgcn_exp2f(s[u][t][r] - mrow[u][r]);
#pragma unroll
            for (int r = 0; r < 8; ++r) lrow[u][r] *= alpha2[u][r];
#pragma unroll
            for (int f = 0; f < 4; ++f)
#pragma unroll
                for (int r = 0; r < 8; ++r)
                    o[u][f][r] *= alpha2[u][r];

            // P (C-layout) -> LDS, f16
#pragma unroll
            for (int t = 0; t < 2; ++t)
#pragma unroll
                for (int r = 0; r < 8; ++r)
                    pw[(u * 16 + r + 8 * half) * 32 + 16 * t + ln] = (_Float16)s[u][t][r];
        }
        asm volatile("" ::: "memory");   // stores before loads (LDS in-order per wave)

        // P A-frags for both M tiles
        v16h pa[2];
#pragma unroll
        for (int u = 0; u < 2; ++u) {
            const _Float16* pr = pw + (u * 16 + ln) * 32 + half * 16;   // 16B aligned
            v8h lo = *(const v8h*)(pr);
            v8h hi = *(const v8h*)(pr + 8);
#pragma unroll
            for (int i = 0; i < 8; ++i) { pa[u][i] = lo[i]; pa[u][i + 8] = hi[i]; }
        }

        // Row sums of P via ones-matrix WMMA: st[r] = sum_k P[row r][k].
#pragma unroll
        for (int u = 0; u < 2; ++u) {
            v8f st = (v8f)0.0f;
            st = __builtin_amdgcn_wmma_f32_16x16x32_f16(
                false, pa[u], false, ones, (short)0, st, false, false);
#pragma unroll
            for (int r = 0; r < 8; ++r) lrow[u][r] += st[r];
        }

        // ---- O += P * V : V frag loaded once, used by both M tiles ----
#pragma unroll
        for (int f = 0; f < 4; ++f) {
            v16h vb;   // B frag: V[j0+k][16f+n], k = half*16 + i, n = ln
            if (F16CACHE) {
                vb = *(const v16h*)(Vt + base + (size_t)(16 * f + ln) * S_
                                    + j0 + half * 16);
            } else {
                const float* vp = V + base + (size_t)(j0 + half * 16) * D_ + 16 * f + ln;
#pragma unroll
                for (int i = 0; i < 16; ++i) vb[i] = (_Float16)vp[(size_t)i * D_];
            }
#pragma unroll
            for (int u = 0; u < 2; ++u)
                o[u][f] = __builtin_amdgcn_wmma_f32_16x16x32_f16(
                    false, pa[u], false, vb, (short)0, o[u][f], false, false);
        }
    };

    // Steady state: fully-unmasked blocks, then the one diagonal block.
    for (int j0 = 0; j0 < q0; j0 += 32)
        block_body(j0, std::false_type{});
    block_body(q0, std::true_type{});

    // ---- Normalize and store ----
#pragma unroll
    for (int u = 0; u < 2; ++u)
#pragma unroll
        for (int f = 0; f < 4; ++f) {
            float* op = Out + base + (size_t)(q0 + u * 16 + 8 * half) * D_ + 16 * f + ln;
#pragma unroll
            for (int r = 0; r < 8; ++r)
                op[(size_t)r * D_] = o[u][f][r] / lrow[u][r];
        }
}

extern "C" void kernel_launch(void* const* d_in, const int* in_sizes, int n_in,
                              void* d_out, int out_size, void* d_ws, size_t ws_size,
                              hipStream_t stream) {
    const float* q = (const float*)d_in[0];
    const float* k = (const float*)d_in[1];
    const float* v = (const float*)d_in[2];
    // d_in[3] is the causal mask; causality is computed analytically in-kernel.
    float* out = (float*)d_out;

    const size_t elems   = (size_t)NHEADS * S_ * D_;            // 4 Mi elements
    const size_t need_ws = 2 * elems * sizeof(unsigned short);  // 16 MB (Kh + Vt)

    const int tasks  = NHEADS * (S_ / QROWS);              // 2048 waves
    const int blocks = tasks / WAVES_PER_BLOCK;            // 256 blocks of 256 threads

    if (ws_size >= need_ws) {
        _Float16* kh = (_Float16*)d_ws;
        _Float16* vt = kh + elems;
        hipLaunchKernelGGL(convert_kv_kernel,
                           dim3((unsigned)(elems / 256)), dim3(256), 0, stream,
                           k, v, kh, vt);
        hipLaunchKernelGGL(ScaledDotProductAttentionModule_1795296330325_kernel<true>,
                           dim3(blocks), dim3(256), 0, stream,
                           q, k, v, (const _Float16*)kh, (const _Float16*)vt, out);
    } else {
        hipLaunchKernelGGL(ScaledDotProductAttentionModule_1795296330325_kernel<false>,
                           dim3(blocks), dim3(256), 0, stream,
                           q, k, v, (const _Float16*)nullptr, (const _Float16*)nullptr, out);
    }
}
